// GerbilizerReducedAttentionNet_67516885893735
// MI455X (gfx1250) — matrix operand order repaired and checked by reading, not compiled, not run on hardware
//
#include <hip/hip_runtime.h>

// ---------------------------------------------------------------------------
// Types for CDNA5 WMMA (wave32): D(16x16,f32) = A(16x32,f16) x B(32x16,f16) + C
// ---------------------------------------------------------------------------
typedef __attribute__((ext_vector_type(16))) _Float16 v16h;
typedef __attribute__((ext_vector_type(8)))  _Float16 v8h;
typedef __attribute__((ext_vector_type(8)))  float    v8f;

// Problem constants (from reference)
#define BB   8
#define SS   2048
#define DD0  512
#define HH0  8
#define DH0  64
#define NB0  32
#define BS0  64
#define FF0  2048
#define MT0  (BB * SS)          // 16384 rows
#define XCH  163760             // NM*T = 80*2047 floats per batch in x

// ---------------------------------------------------------------------------
// CDNA5 async copy: global -> LDS, 16B per lane, tracked by ASYNCcnt.
// ---------------------------------------------------------------------------
__device__ __forceinline__ unsigned lds_off_u32(void* p) {
    return (unsigned)(uintptr_t)(__attribute__((address_space(3))) void*)p;
}

__device__ __forceinline__ void async_copy_b128(unsigned lds_off, const void* base,
                                                unsigned byte_off) {
    asm volatile("global_load_async_to_lds_b128 %0, %1, %2"
                 :: "v"(lds_off), "v"(byte_off), "s"(base)
                 : "memory");
}

__device__ __forceinline__ void wait_asynccnt0() {
    asm volatile("s_wait_asynccnt 0" ::: "memory");
}

// ---------------------------------------------------------------------------
// Fragment loaders from LDS (contiguous 8-half reads)
// A frag (16x32): lane half hl: e<8 -> K = 8*hl + e ; e>=8 -> K = 16 + 8*hl + (e-8)
// B frag (32x16): lane half hl: e<8 -> K = 8*hl + e ; e>=8 -> K = 16 + 8*hl + (e-8)
// ---------------------------------------------------------------------------
__device__ __forceinline__ v16h load_frag_a(const _Float16* rowp, int hl) {
    const v8h p0 = *(const v8h*)(rowp + 8 * hl);
    const v8h p1 = *(const v8h*)(rowp + 16 + 8 * hl);
    v16h a;
#pragma unroll
    for (int i = 0; i < 8; ++i) { a[i] = p0[i]; a[8 + i] = p1[i]; }
    return a;
}

__device__ __forceinline__ v16h load_frag_b(const _Float16* colp, int hl) {
    const v8h p0 = *(const v8h*)(colp + 8 * hl);
    const v8h p1 = *(const v8h*)(colp + 16 + 8 * hl);
    v16h b;
#pragma unroll
    for (int i = 0; i < 8; ++i) { b[i] = p0[i]; b[8 + i] = p1[i]; }
    return b;
}

// ---------------------------------------------------------------------------
// Weight prep: fp32 (K,N) row-major  ->  f16 (N,K) row-major ("Bt" layout)
// ---------------------------------------------------------------------------
__global__ __launch_bounds__(256) void k_transpose_f32_to_f16(
    const float* __restrict__ src, _Float16* __restrict__ dst, int K, int N) {
    size_t i = (size_t)blockIdx.x * 256 + threadIdx.x;
    if (i >= (size_t)K * N) return;
    size_t n = i / (size_t)K, k = i % (size_t)K;
    dst[n * K + k] = (_Float16)src[k * N + n];
}

// ---------------------------------------------------------------------------
// Embedding: h[b,s,:] = (s==0 ? 0 : xr[b,s-1,:] @ w_embed + b_embed) + pos[s,:]
// xr[b,u,j] = x[b*XCH + j*2047 + u]   (reshape+transpose of x)
// ---------------------------------------------------------------------------
__global__ __launch_bounds__(256) void k_embed(
    const float* __restrict__ x, const float* __restrict__ we,
    const float* __restrict__ be, const float* __restrict__ pos,
    float* __restrict__ hf, _Float16* __restrict__ hh) {
    size_t i = (size_t)blockIdx.x * 256 + threadIdx.x;   // over B*S*D
    int d = (int)(i & (DD0 - 1));
    size_t bs = i >> 9;
    int s = (int)(bs & (SS - 1));
    int b = (int)(bs >> 11);
    float acc;
    if (s == 0) {
        acc = pos[d];
    } else {
        acc = be[d] + pos[(size_t)s * DD0 + d];
        const float* xp = x + (size_t)b * XCH + (s - 1);
#pragma unroll 8
        for (int j = 0; j < 80; ++j) acc += xp[(size_t)j * 2047] * we[j * DD0 + d];
    }
    hf[i] = acc;
    hh[i] = (_Float16)acc;
}

// ---------------------------------------------------------------------------
// WMMA GEMM: C[M,N] = A[M,K](f16) @ Bt[N,K](f16)^T + bias, opt relu.
// Block = 256 threads = 8 waves, tile 128x128, K-step 32, double-buffered LDS
// filled by GLOBAL_LOAD_ASYNC_TO_LDS_B128 (ASYNCcnt). Wave w: rows (w&3)*32,
// cols (w>>2)*64 -> 2 A-frags x 4 B-frags -> 8 WMMAs per K-step.
// ---------------------------------------------------------------------------
template <int RELU, int OF32, int OF16>
__global__ __launch_bounds__(256) void k_gemm_f16(
    const _Float16* __restrict__ A, const _Float16* __restrict__ Bt,
    const float* __restrict__ bias, float* __restrict__ Cf,
    _Float16* __restrict__ Ch, int M, int N, int K) {
    __shared__ _Float16 sA[2][128][40];
    __shared__ _Float16 sB[2][128][40];
    const int t = threadIdx.x;
    const int w = t >> 5, l = t & 31, hl = l >> 4, ln = l & 15;
    const int mq = w & 3, nq = w >> 2;
    const int mBase = blockIdx.y * 128, nBase = blockIdx.x * 128;
    const int lr = t >> 1, lc = (t & 1) * 16;   // staging: row, half-offset

    const unsigned gaRow = (unsigned)((((size_t)(mBase + lr)) * K + lc) * 2);
    const unsigned gbRow = (unsigned)((((size_t)(nBase + lr)) * K + lc) * 2);

    // stage K-slice k0 into buffer `buf` (4 x 16B async copies per thread)
    auto stage = [&](int buf, int k0) {
        unsigned la = lds_off_u32(&sA[buf][lr][lc]);
        unsigned lb = lds_off_u32(&sB[buf][lr][lc]);
        unsigned ga = gaRow + (unsigned)(k0 * 2);
        unsigned gb = gbRow + (unsigned)(k0 * 2);
        async_copy_b128(la,      A,  ga);
        async_copy_b128(la + 16, A,  ga + 16);
        async_copy_b128(lb,      Bt, gb);
        async_copy_b128(lb + 16, Bt, gb + 16);
    };

    v8f acc[2][4] = {};

    stage(0, 0);
    wait_asynccnt0();
    __syncthreads();

    int buf = 0;
    for (int k0 = 0; k0 < K; k0 += 32) {
        if (k0 + 32 < K) stage(buf ^ 1, k0 + 32);

        v16h a0 = load_frag_a(&sA[buf][mq * 32 + ln][0], hl);
        v16h a1 = load_frag_a(&sA[buf][mq * 32 + 16 + ln][0], hl);
#pragma unroll
        for (int sn = 0; sn < 4; ++sn) {
            v16h b = load_frag_b(&sB[buf][nq * 64 + sn * 16 + ln][0], hl);
            acc[0][sn] = __builtin_amdgcn_wmma_f32_16x16x32_f16(false, a0, false, b, (short)0, acc[0][sn], false, false);
            acc[1][sn] = __builtin_amdgcn_wmma_f32_16x16x32_f16(false, a1, false, b, (short)0, acc[1][sn], false, false);
        }

        wait_asynccnt0();     // own async copies into buf^1 have landed
        __syncthreads();      // everyone done reading buf / writing buf^1
        buf ^= 1;
    }

#pragma unroll
    for (int sn = 0; sn < 4; ++sn) {
        const int col = nBase + nq * 64 + sn * 16 + ln;
        const float bb = bias[col];
#pragma unroll
        for (int sm = 0; sm < 2; ++sm) {
#pragma unroll
            for (int r = 0; r < 8; ++r) {
                int row = mBase + mq * 32 + sm * 16 + r + 8 * hl;
                float v0 = acc[sm][sn][r] + bb;
                if (RELU) v0 = fmaxf(v0, 0.f);
                size_t o = (size_t)row * N + col;
                if (OF32) Cf[o] = v0;
                if (OF16) Ch[o] = (_Float16)v0;
            }
        }
    }
}

// ---------------------------------------------------------------------------
// Block-sparse flash attention. One block per (b, h, nb); 128 threads = 4 waves.
// Wave w owns query rows [w*16, w*16+16). Iterates C=6 gathered key blocks
// (idx = [0, clip(nb-1..nb+1), rand0, rand1]; duplicate blocks skipped, which
// matches the reference's -1e9 mask under softmax). Online softmax, P bounced
// through LDS to convert C-frag layout -> A-frag layout, o += P @ V via WMMA.
// ---------------------------------------------------------------------------
__global__ __launch_bounds__(128) void k_attention(
    const _Float16* __restrict__ q, const _Float16* __restrict__ k,
    const _Float16* __restrict__ v, const int* __restrict__ rblk,
    _Float16* __restrict__ out) {
    __shared__ _Float16 qs[64][40];       // Q block rows (dh-major)
    __shared__ _Float16 ks[64][40];       // K block rows (dh-major) == Bt layout
    __shared__ _Float16 vt[64][80];       // V transposed: vt[dh][key]
    __shared__ _Float16 pw[4][16][80];    // per-wave P scratch (16 x 64 keys)
    __shared__ int sidx[6];
    __shared__ int sdup[6];

    const int bid = blockIdx.x;
    const int nb = bid & (NB0 - 1);
    const int hh = (bid >> 5) & (HH0 - 1);
    const int b  = bid >> 8;
    const int t = threadIdx.x;
    const int w = t >> 5, l = t & 31, hl = l >> 4, ln = l & 15;

    for (int i = t; i < 512; i += 128) {
        int r = i >> 3, c8 = (i & 7) * 8;
        *(v8h*)&qs[r][c8] =
            *(const v8h*)(q + ((size_t)(b * SS + nb * BS0 + r)) * DD0 + hh * DH0 + c8);
    }
    if (t == 0) {
        sidx[0] = 0;
        for (int j = 0; j < 3; ++j) {
            int vv = nb + j - 1;
            sidx[1 + j] = vv < 0 ? 0 : (vv > NB0 - 1 ? NB0 - 1 : vv);
        }
        sidx[4] = rblk[nb * 2];
        sidx[5] = rblk[nb * 2 + 1];
        for (int c = 0; c < 6; ++c) {
            int d = 0;
            for (int e = 0; e < c; ++e) d |= (sidx[e] == sidx[c]);
            sdup[c] = d;
        }
    }
    __syncthreads();

    float rowm[8], rowl[8];
    v8f acc[4] = {};
#pragma unroll
    for (int r = 0; r < 8; ++r) { rowm[r] = -1e30f; rowl[r] = 0.f; }

    for (int c = 0; c < 6; ++c) {
        if (sdup[c]) continue;                 // block-uniform
        const int jb = sidx[c];
        for (int i = t; i < 512; i += 128) {
            int r = i >> 3, c8 = (i & 7) * 8;
            size_t base = ((size_t)(b * SS + jb * BS0 + r)) * DD0 + hh * DH0 + c8;
            *(v8h*)&ks[r][c8] = *(const v8h*)(k + base);
            v8h vv = *(const v8h*)(v + base);
#pragma unroll
            for (int j = 0; j < 8; ++j) vt[c8 + j][r] = vv[j];
        }
        __syncthreads();

        // scores: S = (Q @ K^T) * 1/sqrt(DH);  4 subtiles of 16 keys, K-dim = 64
        v8f s[4];
#pragma unroll
        for (int nj = 0; nj < 4; ++nj) {
            v8f sc = {};
#pragma unroll
            for (int st = 0; st < 2; ++st) {
                v16h a  = load_frag_a(&qs[w * 16 + ln][st * 32], hl);
                v16h bb = load_frag_b(&ks[nj * 16 + ln][st * 32], hl);
                sc = __builtin_amdgcn_wmma_f32_16x16x32_f16(false, a, false, bb, (short)0, sc, false, false);
            }
#pragma unroll
            for (int r = 0; r < 8; ++r) s[nj][r] = sc[r] * 0.125f;
        }

        // online softmax update (row stats shared by the 16 lanes of each half)
#pragma unroll
        for (int r = 0; r < 8; ++r) {
            float mc = fmaxf(fmaxf(s[0][r], s[1][r]), fmaxf(s[2][r], s[3][r]));
#pragma unroll
            for (int m = 1; m < 16; m <<= 1) mc = fmaxf(mc, __shfl_xor(mc, m, 32));
            float mN = fmaxf(rowm[r], mc);
            float corr = __expf(rowm[r] - mN);
            float lc = 0.f;
#pragma unroll
            for (int nj = 0; nj < 4; ++nj) {
                float pv = __expf(s[nj][r] - mN);
                s[nj][r] = pv;
                lc += pv;
            }
#pragma unroll
            for (int m = 1; m < 16; m <<= 1) lc += __shfl_xor(lc, m, 32);
            rowl[r] = rowl[r] * corr + lc;
            rowm[r] = mN;
#pragma unroll
            for (int dt = 0; dt < 4; ++dt) acc[dt][r] *= corr;
        }

        // P (C-frag layout) -> LDS -> A-frag layout (per-wave region, DS in-order)
#pragma unroll
        for (int nj = 0; nj < 4; ++nj)
#pragma unroll
            for (int r = 0; r < 8; ++r)
                pw[w][r + 8 * hl][nj * 16 + ln] = (_Float16)s[nj][r];

        v16h a0 = load_frag_a(&pw[w][ln][0], hl);    // keys 0..31
        v16h a1 = load_frag_a(&pw[w][ln][32], hl);   // keys 32..63
#pragma unroll
        for (int dt = 0; dt < 4; ++dt) {
            v16h b0 = load_frag_b(&vt[dt * 16 + ln][0], hl);
            v16h b1 = load_frag_b(&vt[dt * 16 + ln][32], hl);
            acc[dt] = __builtin_amdgcn_wmma_f32_16x16x32_f16(false, a0, false, b0, (short)0, acc[dt], false, false);
            acc[dt] = __builtin_amdgcn_wmma_f32_16x16x32_f16(false, a1, false, b1, (short)0, acc[dt], false, false);
        }
        __syncthreads();
    }

#pragma unroll
    for (int dt = 0; dt < 4; ++dt)
#pragma unroll
        for (int r = 0; r < 8; ++r) {
            int row = nb * BS0 + w * 16 + r + 8 * hl;
            out[((size_t)(b * SS + row)) * DD0 + hh * DH0 + dt * 16 + ln] =
                (_Float16)(acc[dt][r] / rowl[r]);
        }
}

// ---------------------------------------------------------------------------
// h = LN(h + add) * g + beta; writes fp32 (in-place) and f16 copy. Block=row.
// ---------------------------------------------------------------------------
__global__ __launch_bounds__(128) void k_add_ln(
    float* __restrict__ h, const float* __restrict__ add,
    const float* __restrict__ g, const float* __restrict__ beta,
    _Float16* __restrict__ hh) {
    __shared__ float sbuf[128];
    const size_t row = blockIdx.x;
    const int t = threadIdx.x;
    float xv[4];
#pragma unroll
    for (int j = 0; j < 4; ++j) {
        size_t idx = row * DD0 + t + j * 128;
        xv[j] = h[idx] + add[idx];
    }
    float s = xv[0] + xv[1] + xv[2] + xv[3];
    sbuf[t] = s;
    __syncthreads();
    for (int st = 64; st > 0; st >>= 1) { if (t < st) sbuf[t] += sbuf[t + st]; __syncthreads(); }
    const float mean = sbuf[0] * (1.f / DD0);
    __syncthreads();
    float qv = 0.f;
#pragma unroll
    for (int j = 0; j < 4; ++j) { float d0 = xv[j] - mean; qv += d0 * d0; }
    sbuf[t] = qv;
    __syncthreads();
    for (int st = 64; st > 0; st >>= 1) { if (t < st) sbuf[t] += sbuf[t + st]; __syncthreads(); }
    const float rs = rsqrtf(sbuf[0] * (1.f / DD0) + 1e-5f);
#pragma unroll
    for (int j = 0; j < 4; ++j) {
        int d = t + j * 128;
        float y = (xv[j] - mean) * rs * g[d] + beta[d];
        size_t idx = row * DD0 + d;
        h[idx] = y;
        hh[idx] = (_Float16)y;
    }
}

// ---------------------------------------------------------------------------
// Head: out[b,:] = tanh(relu(h[b,0,:] @ w_dense + b_dense) @ w_out + b_out)
// ---------------------------------------------------------------------------
__global__ __launch_bounds__(256) void k_head(
    const float* __restrict__ h, const float* __restrict__ wd,
    const float* __restrict__ bd, const float* __restrict__ wo,
    const float* __restrict__ bo, float* __restrict__ out) {
    __shared__ float hrow[DD0];
    __shared__ float dense[1024];
    __shared__ float red[256];
    const int b = blockIdx.x, t = threadIdx.x;
    for (int i = t; i < DD0; i += 256) hrow[i] = h[(size_t)b * SS * DD0 + i];
    __syncthreads();
    for (int i = t; i < 1024; i += 256) {
        float a = bd[i];
        for (int kk = 0; kk < DD0; ++kk) a += hrow[kk] * wd[kk * 1024 + i];
        dense[i] = fmaxf(a, 0.f);
    }
    __syncthreads();
    for (int o = 0; o < 2; ++o) {
        float p = 0.f;
        for (int kk = t; kk < 1024; kk += 256) p += dense[kk] * wo[kk * 2 + o];
        red[t] = p;
        __syncthreads();
        for (int st = 128; st > 0; st >>= 1) { if (t < st) red[t] += red[t + st]; __syncthreads(); }
        if (t == 0) out[b * 2 + o] = tanhf(red[0] + bo[o]);
        __syncthreads();
    }
}

// ---------------------------------------------------------------------------
// Host driver
// ---------------------------------------------------------------------------
extern "C" void kernel_launch(void* const* d_in, const int* in_sizes, int n_in,
                              void* d_out, int out_size, void* d_ws, size_t ws_size,
                              hipStream_t stream) {
    (void)in_sizes; (void)n_in; (void)out_size;
    const float* x       = (const float*)d_in[0];
    const float* w_embed = (const float*)d_in[1];
    const float* b_embed = (const float*)d_in[2];
    const float* pos_emb = (const float*)d_in[3];
    const float* Wq = (const float*)d_in[4];  const float* bq = (const float*)d_in[5];
    const float* Wk = (const float*)d_in[6];  const float* bk = (const float*)d_in[7];
    const float* Wv = (const float*)d_in[8];  const float* bv = (const float*)d_in[9];
    const float* Wo = (const float*)d_in[10]; const float* bo = (const float*)d_in[11];
    const float* ln1_g = (const float*)d_in[12]; const float* ln1_b = (const float*)d_in[13];
    const float* W1 = (const float*)d_in[14]; const float* b1 = (const float*)d_in[15];
    const float* W2 = (const float*)d_in[16]; const float* b2 = (const float*)d_in[17];
    const float* ln2_g = (const float*)d_in[18]; const float* ln2_b = (const float*)d_in[19];
    const float* w_dense = (const float*)d_in[20]; const float* b_dense = (const float*)d_in[21];
    const float* w_out   = (const float*)d_in[22]; const float* b_out   = (const float*)d_in[23];
    const int*   rblk    = (const int*)d_in[24];
    float* outp = (float*)d_out;

    const size_t DDm = (size_t)DD0 * DD0;       // 262144
    const size_t DFF = (size_t)DD0 * FF0;       // 1048576
    char* ws = (char*)d_ws;
    size_t off = 0;
    auto alloc_h = [&](size_t n) { _Float16* p = (_Float16*)(ws + off); off += n * 2; return p; };
    auto alloc_f = [&](size_t n) { float* p = (float*)(ws + off); off += n * 4; return p; };

    _Float16* wqT = alloc_h(4 * DDm);
    _Float16* wkT = alloc_h(4 * DDm);
    _Float16* wvT = alloc_h(4 * DDm);
    _Float16* woT = alloc_h(4 * DDm);
    _Float16* w1T = alloc_h(4 * DFF);
    _Float16* w2T = alloc_h(4 * DFF);
    _Float16* h_h    = alloc_h((size_t)MT0 * DD0);
    _Float16* q_h    = alloc_h((size_t)MT0 * DD0);
    _Float16* k_h    = alloc_h((size_t)MT0 * DD0);
    _Float16* v_h    = alloc_h((size_t)MT0 * DD0);
    _Float16* attn_h = alloc_h((size_t)MT0 * DD0);
    _Float16* ff_h   = alloc_h((size_t)MT0 * FF0);
    float* h_f   = alloc_f((size_t)MT0 * DD0);
    float* tmp_f = alloc_f((size_t)MT0 * DD0);
    if (off > ws_size) return;   // workspace too small: deterministic no-op

    const int gDD  = (int)((DDm + 255) / 256);
    const int gDFF = (int)((DFF + 255) / 256);
    for (int l = 0; l < 4; ++l) {
        k_transpose_f32_to_f16<<<gDD, 256, 0, stream>>>(Wq + l * DDm, wqT + l * DDm, DD0, DD0);
        k_transpose_f32_to_f16<<<gDD, 256, 0, stream>>>(Wk + l * DDm, wkT + l * DDm, DD0, DD0);
        k_transpose_f32_to_f16<<<gDD, 256, 0, stream>>>(Wv + l * DDm, wvT + l * DDm, DD0, DD0);
        k_transpose_f32_to_f16<<<gDD, 256, 0, stream>>>(Wo + l * DDm, woT + l * DDm, DD0, DD0);
        k_transpose_f32_to_f16<<<gDFF, 256, 0, stream>>>(W1 + l * DFF, w1T + l * DFF, DD0, FF0);
        k_transpose_f32_to_f16<<<gDFF, 256, 0, stream>>>(W2 + l * DFF, w2T + l * DFF, FF0, DD0);
    }

    k_embed<<<(MT0 * DD0) / 256, 256, 0, stream>>>(x, w_embed, b_embed, pos_emb, h_f, h_h);

    const dim3 gp(DD0 / 128, MT0 / 128);   // projection GEMMs: N=512
    const dim3 gf(FF0 / 128, MT0 / 128);   // FF1 GEMM: N=2048
    for (int l = 0; l < 4; ++l) {
        k_gemm_f16<0, 0, 1><<<gp, 256, 0, stream>>>(h_h, wqT + l * DDm, bq + l * DD0, nullptr, q_h, MT0, DD0, DD0);
        k_gemm_f16<0, 0, 1><<<gp, 256, 0, stream>>>(h_h, wkT + l * DDm, bk + l * DD0, nullptr, k_h, MT0, DD0, DD0);
        k_gemm_f16<0, 0, 1><<<gp, 256, 0, stream>>>(h_h, wvT + l * DDm, bv + l * DD0, nullptr, v_h, MT0, DD0, DD0);
        k_attention<<<BB * HH0 * NB0, 128, 0, stream>>>(q_h, k_h, v_h, rblk, attn_h);
        k_gemm_f16<0, 1, 0><<<gp, 256, 0, stream>>>(attn_h, woT + l * DDm, bo + l * DD0, tmp_f, nullptr, MT0, DD0, DD0);
        k_add_ln<<<MT0, 128, 0, stream>>>(h_f, tmp_f, ln1_g + l * DD0, ln1_b + l * DD0, h_h);
        k_gemm_f16<1, 0, 1><<<gf, 256, 0, stream>>>(h_h, w1T + l * DFF, b1 + l * FF0, nullptr, ff_h, MT0, FF0, DD0);
        k_gemm_f16<0, 1, 0><<<gp, 256, 0, stream>>>(ff_h, w2T + l * DFF, b2 + l * DD0, tmp_f, nullptr, MT0, DD0, FF0);
        k_add_ln<<<MT0, 128, 0, stream>>>(h_f, tmp_f, ln2_g + l * DD0, ln2_b + l * DD0, h_h);
    }

    k_head<<<BB, 256, 0, stream>>>(h_f, w_dense, b_dense, w_out, b_out, outp);
}
